// DocumentMultiTaskGAT_88708254531847
// MI455X (gfx1250) — matrix operand order repaired and checked by reading, not compiled, not run on hardware
//
#include <hip/hip_runtime.h>
#include <hip/hip_bf16.h>

// ---------------- problem constants ----------------
#define NN 20000
#define EE 320000
#define ET (EE + NN)   // edges + self loops

typedef __attribute__((ext_vector_type(16))) __bf16 v16bf;
typedef __attribute__((ext_vector_type(8)))  __bf16 v8bf;
typedef __attribute__((ext_vector_type(8)))  float  v8f;

// ---------------- small device helpers ----------------
__device__ __forceinline__ float eluf(float v) {
    return v > 0.f ? v : (__expf(v) - 1.f);
}

__device__ __forceinline__ void atomicMaxF(float* addr, float val) {
    unsigned int* ai = reinterpret_cast<unsigned int*>(addr);
    unsigned int old = *ai;
    while (__uint_as_float(old) < val) {
        unsigned int assumed = old;
        old = atomicCAS(ai, assumed, __float_as_uint(val));
        if (old == assumed) break;
    }
}

// ---------------- generic fill ----------------
__global__ void fill_kernel(float* __restrict__ p, float v, int n) {
    int i = blockIdx.x * blockDim.x + threadIdx.x;
    if (i < n) p[i] = v;
}

// ---------------- weight pack: f32 [din,dout] -> bf16 B-operand layout ----------------
// GEMM reads: Wp[ ((tn*nk + kk)*32 + lane)*16 + i ], element = B[k][n],
// k = kk*32 + (lane>>4)*16 + i, n = tn*16 + (lane&15)   (contiguous k order, same as the A reads)
__global__ void pack_w_kernel(const float* __restrict__ W, __hip_bfloat16* __restrict__ out,
                              int din, int dout, int din_p, int dout_p) {
    int idx = blockIdx.x * blockDim.x + threadIdx.x;
    int total = din_p * dout_p;
    if (idx >= total) return;
    int nk = din_p >> 5;
    int i    = idx & 15;
    int lane = (idx >> 4) & 31;
    int blk  = idx >> 9;          // tn*nk + kk
    int kk   = blk % nk;
    int tn   = blk / nk;
    int k = kk * 32 + ((lane >> 4) << 4) + i;
    int n = tn * 16 + (lane & 15);
    float v = (k < din && n < dout) ? W[(size_t)k * dout + n] : 0.f;
    out[idx] = __float2bfloat16(v);
}

// ---------------- activation pack: f32 [rows,din] -> bf16 [rows,din_p] (zero pad) ----------------
__global__ void pack_a_kernel(const float* __restrict__ in, __hip_bfloat16* __restrict__ out,
                              int rows, int din, int din_p) {
    int idx = blockIdx.x * blockDim.x + threadIdx.x;
    if (idx >= rows * din_p) return;
    int r = idx / din_p;
    int c = idx - r * din_p;
    float v = (c < din) ? in[(size_t)r * din + c] : 0.f;
    out[idx] = __float2bfloat16(v);
}

// ---------------- gate input: bf16 [N,288] = [ z(256) | yolo(11) | pad ] ----------------
__global__ void build_gate_in_kernel(const float* __restrict__ z, const float* __restrict__ yolo,
                                     __hip_bfloat16* __restrict__ out) {
    int idx = blockIdx.x * blockDim.x + threadIdx.x;
    if (idx >= NN * 288) return;
    int r = idx / 288;
    int c = idx - r * 288;
    float v = 0.f;
    if (c < 256)       v = z[(size_t)r * 256 + c];
    else if (c < 267)  v = yolo[(size_t)r * 11 + (c - 256)];
    out[idx] = __float2bfloat16(v);
}

// ---------------- WMMA GEMM: one wave -> 16x32 output slab (two 16x16 tiles) ----------------
// Block (32 x 4 waves): all waves share the same 16-row A slab -> staged once in LDS
// (4x less global A gather traffic, fragment reads become ds_load_b128).
// Per wave: shared A fragment, two B tiles, two independent accumulators, and an explicit
// 2-stage software pipeline on the B fragments so loads of step k+1 overlap the WMMAs of k.
#define LOADB(dst, p) { dst.h[0] = *reinterpret_cast<const v8bf*>(p); \
                        dst.h[1] = *reinterpret_cast<const v8bf*>((p) + 8); }
#define LOADA(dst, p) { dst.h[0] = *reinterpret_cast<const v8bf*>(p); \
                        dst.h[1] = *reinterpret_cast<const v8bf*>((p) + 8); }

__global__ void gemm_bf16_kernel(const __hip_bfloat16* __restrict__ A, int lda,
                                 const __hip_bfloat16* __restrict__ Wp,
                                 const float* __restrict__ bias,
                                 float* __restrict__ C, int ldc, int coloff,
                                 int n_pairs, int nk, int act) {
    extern __shared__ __align__(16) char smem[];   // 16 * lda * 2 bytes
    int tm   = blockIdx.x;
    int lane = threadIdx.x;
    int wy   = threadIdx.y;

    // cooperative copy of A slab [16, lda] -> LDS (16B chunks)
    {
        const uint4* gsrc = reinterpret_cast<const uint4*>(A + (size_t)tm * 16 * lda);
        uint4* ldst = reinterpret_cast<uint4*>(smem);
        int nchunks = (16 * lda) >> 3;             // 8 bf16 per 16B chunk
        for (int t = wy * 32 + lane; t < nchunks; t += blockDim.y * 32) ldst[t] = gsrc[t];
    }
    __syncthreads();

    int tp = blockIdx.y * blockDim.y + wy;         // which 32-col pair
    if (tp < n_pairs) {
        int half = lane >> 4;
        int nl   = lane & 15;
        const __bf16* sA = reinterpret_cast<const __bf16*>(smem);
        const __bf16* Bb = reinterpret_cast<const __bf16*>(Wp);

        // A: row = lane&15; elems 0..15 -> k = kbase + half*16 + j
        const __bf16* ap  = sA + (size_t)nl * lda + half * 16;
        // B: lane-contiguous 32B per k-step, stride 512 elements per kk
        const __bf16* bp0 = Bb + ((size_t)(2 * tp) * nk * 32 + lane) * 16;
        const __bf16* bp1 = Bb + ((size_t)(2 * tp + 1) * nk * 32 + lane) * 16;

        union U { v16bf v; v8bf h[2]; };
        U a0, a1, b00, b10, b01, b11;
        v8f c0 = {};
        v8f c1 = {};

        // prologue: step 0
        LOADA(a0, ap);  LOADB(b00, bp0);  LOADB(b10, bp1);
        ap += 32; bp0 += 512; bp1 += 512;
        int kk = 1;
        for (; kk + 1 < nk; kk += 2) {
            LOADA(a1, ap);  LOADB(b01, bp0);  LOADB(b11, bp1);
            ap += 32; bp0 += 512; bp1 += 512;
            c0 = __builtin_amdgcn_wmma_f32_16x16x32_bf16(false, a0.v, false, b00.v,
                                                         (short)0, c0, false, false);
            c1 = __builtin_amdgcn_wmma_f32_16x16x32_bf16(false, a0.v, false, b10.v,
                                                         (short)0, c1, false, false);
            LOADA(a0, ap);  LOADB(b00, bp0);  LOADB(b10, bp1);
            ap += 32; bp0 += 512; bp1 += 512;
            c0 = __builtin_amdgcn_wmma_f32_16x16x32_bf16(false, a1.v, false, b01.v,
                                                         (short)0, c0, false, false);
            c1 = __builtin_amdgcn_wmma_f32_16x16x32_bf16(false, a1.v, false, b11.v,
                                                         (short)0, c1, false, false);
        }
        if (kk < nk) {          // one trailing step still to load
            LOADA(a1, ap);  LOADB(b01, bp0);  LOADB(b11, bp1);
            c0 = __builtin_amdgcn_wmma_f32_16x16x32_bf16(false, a0.v, false, b00.v,
                                                         (short)0, c0, false, false);
            c1 = __builtin_amdgcn_wmma_f32_16x16x32_bf16(false, a0.v, false, b10.v,
                                                         (short)0, c1, false, false);
            c0 = __builtin_amdgcn_wmma_f32_16x16x32_bf16(false, a1.v, false, b01.v,
                                                         (short)0, c0, false, false);
            c1 = __builtin_amdgcn_wmma_f32_16x16x32_bf16(false, a1.v, false, b11.v,
                                                         (short)0, c1, false, false);
        } else {                // nk was odd-consumed: just finish step kk-1
            c0 = __builtin_amdgcn_wmma_f32_16x16x32_bf16(false, a0.v, false, b00.v,
                                                         (short)0, c0, false, false);
            c1 = __builtin_amdgcn_wmma_f32_16x16x32_bf16(false, a0.v, false, b10.v,
                                                         (short)0, c1, false, false);
        }

        // C/D layout: VGPR r -> M = r + half*8 ; N = lane&15
        int ncol0 = tp * 32 + nl;
        int ncol1 = ncol0 + 16;
        float bb0 = bias ? bias[ncol0] : 0.f;
        float bb1 = bias ? bias[ncol1] : 0.f;
#pragma unroll
        for (int r = 0; r < 8; ++r) {
            int m = tm * 16 + r + half * 8;
            float v0 = c0[r] + bb0;
            float v1 = c1[r] + bb1;
            if (act == 1) { v0 = eluf(v0); v1 = eluf(v1); }
            float* crow = C + (size_t)m * ldc + coloff;
            crow[ncol0] = v0;
            crow[ncol1] = v1;
        }
    }
}

// ---------------- GATv2 edge kernels ----------------
__global__ void edge_logits_kernel(const int* __restrict__ src, const int* __restrict__ dst,
                                   const float* __restrict__ xl, const float* __restrict__ xr,
                                   const float* __restrict__ att, float* __restrict__ alpha,
                                   float* __restrict__ mbuf, int H, int C) {
    int idx = blockIdx.x * blockDim.x + threadIdx.x;
    if (idx >= ET * H) return;
    int e = idx / H;
    int h = idx - e * H;
    int s = (e < EE) ? src[e] : (e - EE);
    int d = (e < EE) ? dst[e] : (e - EE);
    const float* pl = xl + (size_t)s * 256 + h * C;
    const float* pr = xr + (size_t)d * 256 + h * C;
    const float* pa = att + h * C;
    float acc = 0.f;
    for (int c = 0; c < C; ++c) {
        float t = pl[c] + pr[c];
        t = t > 0.f ? t : 0.2f * t;      // leaky_relu, slope 0.2
        acc += pa[c] * t;
    }
    alpha[idx] = acc;
    atomicMaxF(&mbuf[d * H + h], acc);
}

__global__ void edge_exp_kernel(const int* __restrict__ dst,
                                float* __restrict__ alpha, const float* __restrict__ mbuf,
                                float* __restrict__ sbuf, int H) {
    int idx = blockIdx.x * blockDim.x + threadIdx.x;
    if (idx >= ET * H) return;
    int e = idx / H;
    int h = idx - e * H;
    int d = (e < EE) ? dst[e] : (e - EE);
    float a = __expf(alpha[idx] - mbuf[d * H + h]);
    alpha[idx] = a;
    atomicAdd(&sbuf[d * H + h], a);
}

// one thread = 4 consecutive channels (same head since C % 4 == 0): one float4 gather,
// one alpha/s lookup, 4 atomic adds into the L2-resident accumulator
__global__ void edge_scatter_kernel(const int* __restrict__ src, const int* __restrict__ dst,
                                    const float* __restrict__ xl, const float* __restrict__ alpha,
                                    const float* __restrict__ sbuf, float* __restrict__ acc,
                                    int H, int C) {
    int idx = blockIdx.x * blockDim.x + threadIdx.x;   // ET*64 threads
    if (idx >= ET * 64) return;
    int e  = idx >> 6;
    int r4 = (idx & 63) << 2;          // channel group within [0,256)
    int h  = r4 / C;
    int s = (e < EE) ? src[e] : (e - EE);
    int d = (e < EE) ? dst[e] : (e - EE);
    float w = alpha[(size_t)e * H + h] / sbuf[d * H + h];
    const float4 xv = *reinterpret_cast<const float4*>(xl + (size_t)s * 256 + r4);
    float* ap = acc + (size_t)d * 256 + r4;
    atomicAdd(ap + 0, xv.x * w);
    atomicAdd(ap + 1, xv.y * w);
    atomicAdd(ap + 2, xv.z * w);
    atomicAdd(ap + 3, xv.w * w);
}

__global__ void finalize_kernel(const float* __restrict__ acc, const float* __restrict__ bias,
                                const float* __restrict__ xres, float* __restrict__ out,
                                int residual) {
    int idx = blockIdx.x * blockDim.x + threadIdx.x;
    if (idx >= NN * 256) return;
    int col = idx & 255;
    float v = eluf(acc[idx] + bias[col]);
    if (residual) v += xres[idx];
    out[idx] = v;
}

// ---------------- heads ----------------
__global__ void bn_elu_kernel(float* __restrict__ h, const float* __restrict__ gamma,
                              const float* __restrict__ beta, const float* __restrict__ mean,
                              const float* __restrict__ var) {
    int idx = blockIdx.x * blockDim.x + threadIdx.x;
    if (idx >= NN * 128) return;
    int c = idx & 127;
    float v = (h[idx] - mean[c]) * rsqrtf(var[c] + 1e-5f) * gamma[c] + beta[c];
    h[idx] = eluf(v);
}

__global__ void cls2_kernel(const float* __restrict__ h, const float* __restrict__ W,
                            const float* __restrict__ b, float* __restrict__ out) {
    int idx = blockIdx.x * blockDim.x + threadIdx.x;   // N*8
    if (idx >= NN * 8) return;
    int n = idx >> 3;
    int o = idx & 7;
    const float* hp = h + (size_t)n * 128;
    float acc = b[o];
    for (int j = 0; j < 128; ++j) acc += hp[j] * W[j * 8 + o];
    out[idx] = acc;
}

__global__ void gate2_kernel(const float* __restrict__ g, const float* __restrict__ W,
                             const float* __restrict__ b, float* __restrict__ out) {
    int n = blockIdx.x * blockDim.x + threadIdx.x;
    if (n >= NN) return;
    const float* gp = g + (size_t)n * 64;
    float acc = b[0];
    for (int j = 0; j < 64; ++j) acc += gp[j] * W[j];
    out[n] = 1.f / (1.f + __expf(-acc));
}

// ---------------- workspace layout (bytes) ----------------
static constexpr size_t OFF_X     = 0;                               // N*256*4
static constexpr size_t OFF_XL    = OFF_X     + 20480000;
static constexpr size_t OFF_XR    = OFF_XL    + 20480000;
static constexpr size_t OFF_ACC   = OFF_XR    + 20480000;
static constexpr size_t OFF_XB    = OFF_ACC   + 20480000;            // N*256*2 bf16
static constexpr size_t OFF_AB    = OFF_XB    + 10240000;            // N*416*2 bf16 staging
static constexpr size_t OFF_ALPHA = OFF_AB    + 16640000;            // ET*8*4
static constexpr size_t OFF_M     = OFF_ALPHA + 10880000;            // N*8*4
static constexpr size_t OFF_S     = OFF_M     + 640000;
static constexpr size_t OFF_G64   = OFF_S     + 640000;              // N*64*4
static constexpr size_t OFF_H128  = OFF_G64   + 5120000;             // N*128*4
static constexpr size_t OFF_WP    = OFF_H128  + 10240000;            // packed weights

// packed-weight sub-offsets (bf16 elements)
static constexpr size_t PW_GEOM = 0;        // 32*64
static constexpr size_t PW_YOLO = 2048;     // 32*32
static constexpr size_t PW_TEXT = 3072;     // 416*160
static constexpr size_t PW_L1   = 69632;    // 256*256
static constexpr size_t PW_R1   = 135168;
static constexpr size_t PW_L2   = 200704;
static constexpr size_t PW_R2   = 266240;
static constexpr size_t PW_L3   = 331776;
static constexpr size_t PW_R3   = 397312;
static constexpr size_t PW_G1   = 462848;   // 288*64
static constexpr size_t PW_C1   = 481280;   // 256*128

static inline int nb(long long n) { return (int)((n + 255) / 256); }

extern "C" void kernel_launch(void* const* d_in, const int* in_sizes, int n_in,
                              void* d_out, int out_size, void* d_ws, size_t ws_size,
                              hipStream_t stream) {
    // -------- inputs (setup_inputs dict order, params flattened in insertion order) --------
    const float* feat_geom = (const float*)d_in[0];
    const float* feat_yolo = (const float*)d_in[1];
    const float* feat_text = (const float*)d_in[2];
    const int*   edge_idx  = (const int*)d_in[3];   // [2,E]
    const float* geom_W = (const float*)d_in[4];  const float* geom_b = (const float*)d_in[5];
    const float* yolo_W = (const float*)d_in[6];  const float* yolo_b = (const float*)d_in[7];
    const float* text_W = (const float*)d_in[8];  const float* text_b = (const float*)d_in[9];
    const float* cWl[3] = { (const float*)d_in[10], (const float*)d_in[16], (const float*)d_in[22] };
    const float* cbl[3] = { (const float*)d_in[11], (const float*)d_in[17], (const float*)d_in[23] };
    const float* cWr[3] = { (const float*)d_in[12], (const float*)d_in[18], (const float*)d_in[24] };
    const float* cbr[3] = { (const float*)d_in[13], (const float*)d_in[19], (const float*)d_in[25] };
    const float* catt[3]= { (const float*)d_in[14], (const float*)d_in[20], (const float*)d_in[26] };
    const float* cbias[3]={ (const float*)d_in[15], (const float*)d_in[21], (const float*)d_in[27] };
    const float* gate_W1=(const float*)d_in[28]; const float* gate_b1=(const float*)d_in[29];
    const float* gate_W2=(const float*)d_in[30]; const float* gate_b2=(const float*)d_in[31];
    const float* cls_W1 =(const float*)d_in[32]; const float* cls_b1 =(const float*)d_in[33];
    const float* bn_g=(const float*)d_in[34]; const float* bn_b=(const float*)d_in[35];
    const float* bn_m=(const float*)d_in[36]; const float* bn_v=(const float*)d_in[37];
    const float* cls_W2 =(const float*)d_in[38]; const float* cls_b2 =(const float*)d_in[39];

    const int* esrc = edge_idx;
    const int* edst = edge_idx + EE;

    char* ws = (char*)d_ws;
    float* X    = (float*)(ws + OFF_X);
    float* XL   = (float*)(ws + OFF_XL);
    float* XR   = (float*)(ws + OFF_XR);
    float* ACC  = (float*)(ws + OFF_ACC);
    __hip_bfloat16* XB = (__hip_bfloat16*)(ws + OFF_XB);
    __hip_bfloat16* AB = (__hip_bfloat16*)(ws + OFF_AB);
    float* ALPHA= (float*)(ws + OFF_ALPHA);
    float* MB   = (float*)(ws + OFF_M);
    float* SB   = (float*)(ws + OFF_S);
    float* G64  = (float*)(ws + OFF_G64);
    float* H128 = (float*)(ws + OFF_H128);
    __hip_bfloat16* WP = (__hip_bfloat16*)(ws + OFF_WP);

    float* Z        = (float*)d_out;                 // [N,256]
    float* LOGITS   = Z + (size_t)NN * 256;          // [N,8]
    float* GATE     = LOGITS + (size_t)NN * 8;       // [N,1]

    const int TPB = 256;
    dim3 gblock(32, 4);

    // -------- pack all weights to bf16 B-operand layout --------
    auto packw = [&](const float* W, size_t off, int din, int dout, int dinp, int doutp) {
        pack_w_kernel<<<nb((long long)dinp * doutp), TPB, 0, stream>>>(W, WP + off, din, dout, dinp, doutp);
    };
    packw(geom_W,  PW_GEOM, 11, 64, 32, 64);
    packw(yolo_W,  PW_YOLO, 11, 32, 32, 32);
    packw(text_W,  PW_TEXT, 389, 160, 416, 160);
    packw(cWl[0], PW_L1, 256, 256, 256, 256);  packw(cWr[0], PW_R1, 256, 256, 256, 256);
    packw(cWl[1], PW_L2, 256, 256, 256, 256);  packw(cWr[1], PW_R2, 256, 256, 256, 256);
    packw(cWl[2], PW_L3, 256, 256, 256, 256);  packw(cWr[2], PW_R3, 256, 256, 256, 256);
    packw(gate_W1, PW_G1, 267, 64, 288, 64);
    packw(cls_W1,  PW_C1, 256, 128, 256, 128);

    // npairs = ntiles/2 (all output widths are multiples of 32); smem = 16*lda*2 bytes
    auto gemm = [&](const __hip_bfloat16* A, int lda, size_t wpoff, const float* bias,
                    float* C, int ldc, int coloff, int npairs, int nk, int act) {
        dim3 grid(NN / 16, (npairs + 3) / 4);
        size_t smem = (size_t)16 * lda * 2;
        gemm_bf16_kernel<<<grid, gblock, smem, stream>>>(A, lda, WP + wpoff, bias, C, ldc,
                                                         coloff, npairs, nk, act);
    };

    // -------- input projections (ELU fused), write into column slices of X --------
    pack_a_kernel<<<nb((long long)NN * 32), TPB, 0, stream>>>(feat_geom, AB, NN, 11, 32);
    gemm(AB, 32, PW_GEOM, geom_b, X, 256, 0, 2, 1, 1);
    pack_a_kernel<<<nb((long long)NN * 32), TPB, 0, stream>>>(feat_yolo, AB, NN, 11, 32);
    gemm(AB, 32, PW_YOLO, yolo_b, X, 256, 64, 1, 1, 1);
    pack_a_kernel<<<nb((long long)NN * 416), TPB, 0, stream>>>(feat_text, AB, NN, 389, 416);
    gemm(AB, 416, PW_TEXT, text_b, X, 256, 96, 5, 13, 1);

    // -------- 3 GATv2 layers --------
    const size_t pwl[3] = { PW_L1, PW_L2, PW_L3 };
    const size_t pwr[3] = { PW_R1, PW_R2, PW_R3 };
    const int heads[3] = { 4, 4, 8 };
    for (int layer = 0; layer < 3; ++layer) {
        int H = heads[layer], C = 256 / H;
        pack_a_kernel<<<nb((long long)NN * 256), TPB, 0, stream>>>(X, XB, NN, 256, 256);
        gemm(XB, 256, pwl[layer], cbl[layer], XL, 256, 0, 8, 8, 0);
        gemm(XB, 256, pwr[layer], cbr[layer], XR, 256, 0, 8, 8, 0);

        fill_kernel<<<nb(NN * 8), TPB, 0, stream>>>(MB, -1e30f, NN * H);
        fill_kernel<<<nb(NN * 8), TPB, 0, stream>>>(SB, 0.f, NN * H);
        fill_kernel<<<nb((long long)NN * 256), TPB, 0, stream>>>(ACC, 0.f, NN * 256);

        edge_logits_kernel<<<nb((long long)ET * H), TPB, 0, stream>>>(esrc, edst, XL, XR,
                                                                      catt[layer], ALPHA, MB, H, C);
        edge_exp_kernel<<<nb((long long)ET * H), TPB, 0, stream>>>(edst, ALPHA, MB, SB, H);
        edge_scatter_kernel<<<nb((long long)ET * 64), TPB, 0, stream>>>(esrc, edst, XL, ALPHA,
                                                                        SB, ACC, H, C);
        if (layer < 2)
            finalize_kernel<<<nb((long long)NN * 256), TPB, 0, stream>>>(ACC, cbias[layer], X, X, 1);
        else
            finalize_kernel<<<nb((long long)NN * 256), TPB, 0, stream>>>(ACC, cbias[layer], X, Z, 0);
    }

    // -------- gate head: elu([z|yolo] @ W1 + b1) -> sigmoid(@W2 + b2) --------
    build_gate_in_kernel<<<nb((long long)NN * 288), TPB, 0, stream>>>(Z, feat_yolo, AB);
    gemm(AB, 288, PW_G1, gate_b1, G64, 64, 0, 2, 9, 1);
    gate2_kernel<<<nb(NN), TPB, 0, stream>>>(G64, gate_W2, gate_b2, GATE);

    // -------- classifier head: z @ W1 + b1 -> BN -> ELU -> @W2 + b2 --------
    pack_a_kernel<<<nb((long long)NN * 256), TPB, 0, stream>>>(Z, XB, NN, 256, 256);
    gemm(XB, 256, PW_C1, cls_b1, H128, 128, 0, 4, 8, 0);
    bn_elu_kernel<<<nb((long long)NN * 128), TPB, 0, stream>>>(H128, bn_g, bn_b, bn_m, bn_v);
    cls2_kernel<<<nb((long long)NN * 8), TPB, 0, stream>>>(H128, cls_W2, cls_b2, LOGITS);

    (void)in_sizes; (void)n_in; (void)out_size; (void)ws_size;
}
